// APPNP2_25159918420551
// MI455X (gfx1250) — compile-verified, matrix operand order checked by reading, not run on hardware
//
#include <hip/hip_runtime.h>

// ---------------------------------------------------------------------------
// Types for CDNA5 WMMA (wave32, 16x16x32 bf16 -> f32)
// ---------------------------------------------------------------------------
typedef __attribute__((ext_vector_type(16))) __bf16 v16bf;
typedef __attribute__((ext_vector_type(8)))  __bf16 v8bf;
typedef __attribute__((ext_vector_type(8)))  float  v8f;

__device__ __forceinline__ __bf16 f2bf(float f) {
    union { float f; unsigned u; } cv; cv.f = f;
    unsigned r = (cv.u + 0x7FFFu + ((cv.u >> 16) & 1u)) >> 16;   // round-nearest-even
    union { unsigned short s; __bf16 b; } o; o.s = (unsigned short)r;
    return o.b;
}

// one-shot fp32 -> bf16 weight conversion (weights are tiny, L2-resident)
__global__ void wconvert_kernel(const float* __restrict__ W, __bf16* __restrict__ Wb, int n)
{
    int i = blockIdx.x * blockDim.x + threadIdx.x;
    if (i < n) Wb[i] = f2bf(W[i]);
}

// ---------------------------------------------------------------------------
// WMMA MLP layer:  Out[M,Nc] = act(A[M,K] * W[Nc,K]^T + bias)
// Block = 8 waves = 256 threads. Wave tile = 16M x (16*NT)N, block = 128M x 16NT N.
// One A fragment feeds NT back-to-back WMMAs -> NT x arithmetic intensity on A.
// W staging is register double-buffered: next K-tile global loads overlap the
// current tile's WMMAs; iteration top is just ds_store of prefetched registers.
// ---------------------------------------------------------------------------
template<int NT, bool A_BF16, bool OUT_BF16, bool RELU>
__global__ __launch_bounds__(256) void wmma_mlp_layer(
    const void* __restrict__ Aptr, const __bf16* __restrict__ Wb,
    const float* __restrict__ bias, void* __restrict__ Outp,
    int M, int K, int Nc)
{
    __shared__ __align__(16) __bf16 Bs[16 * NT * 40];  // stride 40 elems = 80B (16B-aligned rows)

    const int tid  = threadIdx.x;
    const int lane = tid & 31;
    const int wave = tid >> 5;
    const int half = lane >> 4;      // K-half selector for A/B fragments
    const int ln   = lane & 15;      // M-row (A) / N-col (B, C/D)

    const int mBase = (blockIdx.x * 8 + wave) * 16;
    const int nBase = blockIdx.y * (16 * NT);

    int mRow = mBase + ln;
    if (mRow >= M) mRow = M - 1;     // clamp loads; stores guarded below

    const float*  Af = (const float*)Aptr;
    const __bf16* Ab = (const __bf16*)Aptr;

    // ---- staging descriptors: fixed chunks per thread (compile-time count)
    constexpr int ITER = (16 * NT * 4) / 256;   // 8-elem chunks/thread: 2 (NT=8), 1 (NT=4)
    const __bf16* wsrc[ITER];                    // per-chunk global source (k0 = 0)
    int           sdst[ITER];                    // per-chunk LDS element offset
    v8bf          sreg[ITER];                    // in-flight staging registers
#pragma unroll
    for (int i = 0; i < ITER; ++i) {
        const int c   = tid + i * 256;
        const int row = c >> 2;
        const int cb  = (c & 3) * 8;
        wsrc[i] = Wb + (size_t)(nBase + row) * K + cb;
        sdst[i] = row * 40 + cb;
        sreg[i] = *(const v8bf*)(wsrc[i]);       // preload K-tile 0
    }

    v8f acc[NT] = {};

    for (int k0 = 0; k0 < K; k0 += 32) {
        __syncthreads();
#pragma unroll
        for (int i = 0; i < ITER; ++i)           // commit prefetched W tile to LDS
            *(v8bf*)(&Bs[sdst[i]]) = sreg[i];
        __syncthreads();

        if (k0 + 32 < K) {
#pragma unroll
            for (int i = 0; i < ITER; ++i)       // start next W tile (hidden by WMMAs)
                sreg[i] = *(const v8bf*)(wsrc[i] + k0 + 32);
        }

        // ---- A fragment: row mRow, K runs [k0+8h, +8) and [k0+16+8h, +8)
        v16bf a;
        if (A_BF16) {
            const __bf16* ap = Ab + (size_t)mRow * K + k0 + 8 * half;
            v8bf lo = *(const v8bf*)(ap);
            v8bf hi = *(const v8bf*)(ap + 16);
#pragma unroll
            for (int i = 0; i < 8; ++i) { a[i] = lo[i]; a[i + 8] = hi[i]; }
        } else {
            const float* ap = Af + (size_t)mRow * K + k0 + 8 * half;
            float4 a0 = *(const float4*)(ap);
            float4 a1 = *(const float4*)(ap + 4);
            float4 a2 = *(const float4*)(ap + 16);
            float4 a3 = *(const float4*)(ap + 20);
            a[0]  = f2bf(a0.x); a[1]  = f2bf(a0.y); a[2]  = f2bf(a0.z); a[3]  = f2bf(a0.w);
            a[4]  = f2bf(a1.x); a[5]  = f2bf(a1.y); a[6]  = f2bf(a1.z); a[7]  = f2bf(a1.w);
            a[8]  = f2bf(a2.x); a[9]  = f2bf(a2.y); a[10] = f2bf(a2.z); a[11] = f2bf(a2.w);
            a[12] = f2bf(a3.x); a[13] = f2bf(a3.y); a[14] = f2bf(a3.z); a[15] = f2bf(a3.w);
        }

        // prefetch next A K-tile (global_prefetch_b8, speculative)
        if (k0 + 32 < K) {
            if (A_BF16) __builtin_prefetch(Ab + (size_t)mRow * K + k0 + 32, 0, 3);
            else        __builtin_prefetch(Af + (size_t)mRow * K + k0 + 32, 0, 3);
        }

        // ---- NT WMMAs sharing one A fragment
#pragma unroll
        for (int t = 0; t < NT; ++t) {
            v8bf blo = *(const v8bf*)(&Bs[(16 * t + ln) * 40 + 8 * half]);
            v8bf bhi = *(const v8bf*)(&Bs[(16 * t + ln) * 40 + 16 + 8 * half]);
            v16bf b;
#pragma unroll
            for (int i = 0; i < 8; ++i) { b[i] = blo[i]; b[i + 8] = bhi[i]; }
            acc[t] = __builtin_amdgcn_wmma_f32_16x16x32_bf16(
                         false, a, false, b, (short)0, acc[t], false, false);
        }
    }

    // ---- epilogue: bias + optional relu + store (fast path for full tiles)
    float bv[NT];
#pragma unroll
    for (int t = 0; t < NT; ++t) bv[t] = bias[nBase + 16 * t + ln];

    const int mTop = mBase + 8 * half;   // first of the 8 rows this lane writes
    if (mBase + 16 <= M) {               // full tile: no per-store guards
#pragma unroll
        for (int t = 0; t < NT; ++t) {
            size_t o = (size_t)mTop * Nc + nBase + 16 * t + ln;
#pragma unroll
            for (int v = 0; v < 8; ++v, o += Nc) {
                float r = acc[t][v] + bv[t];
                if (RELU) r = fmaxf(r, 0.0f);
                if (OUT_BF16) ((__bf16*)Outp)[o] = f2bf(r);
                else          ((float*) Outp)[o] = r;
            }
        }
    } else {
#pragma unroll
        for (int t = 0; t < NT; ++t) {
            size_t o = (size_t)mTop * Nc + nBase + 16 * t + ln;
#pragma unroll
            for (int v = 0; v < 8; ++v, o += Nc) {
                if (mTop + v < M) {
                    float r = acc[t][v] + bv[t];
                    if (RELU) r = fmaxf(r, 0.0f);
                    if (OUT_BF16) ((__bf16*)Outp)[o] = f2bf(r);
                    else          ((float*) Outp)[o] = r;
                }
            }
        }
    }
}

// ---------------------------------------------------------------------------
// APPNP propagation kernels
// ---------------------------------------------------------------------------
__global__ void deg_kernel(const int* __restrict__ src, const int* __restrict__ dst,
                           float* __restrict__ dout, float* __restrict__ din, int E)
{
    int i = blockIdx.x * blockDim.x + threadIdx.x;
    if (i >= E) return;
    atomicAdd(&dout[src[i]], 1.0f);
    atomicAdd(&din[dst[i]], 1.0f);
}

__global__ void norm_kernel(const float* __restrict__ dout, const float* __restrict__ din,
                            float* __restrict__ ns, float* __restrict__ nd, int Nn)
{
    int i = blockIdx.x * blockDim.x + threadIdx.x;
    if (i >= Nn) return;
    ns[i] = rsqrtf(fmaxf(dout[i], 1.0f));
    nd[i] = 0.9f * rsqrtf(fmaxf(din[i], 1.0f));   // (1-ALPHA) folded into dst norm
}

// one wave32 per edge; lane handles float2 -> 64 features per edge.
// agg (25.6 MB) is L2-resident (192 MB L2) so the f32 atomics stay on-chip.
__global__ void scatter_kernel(const float* __restrict__ h, const int* __restrict__ src,
                               const int* __restrict__ dst, const float* __restrict__ ns,
                               float* __restrict__ agg, int E, int C)
{
    int gid = blockIdx.x * blockDim.x + threadIdx.x;
    int e = gid >> 5;
    if (e >= E) return;
    int lane = gid & 31;
    int s = src[e], d = dst[e];
    float sc = ns[s];
    const float* hp = h   + (size_t)s * C;
    float*       ap = agg + (size_t)d * C;
    for (int j = lane * 2; j < C; j += 64) {
        float2 v = *(const float2*)(hp + j);
        atomicAdd(ap + j,     v.x * sc);
        atomicAdd(ap + j + 1, v.y * sc);
    }
}

// h_out = agg * norm_dst(incl. 1-a) + a*h0 ; also clears agg for next step
__global__ void update_kernel(float* __restrict__ agg, const float* __restrict__ h0,
                              const float* __restrict__ nd, float* __restrict__ hout,
                              int Nn, int C)
{
    int gid = blockIdx.x * blockDim.x + threadIdx.x;   // one float2 per thread
    int total = Nn * (C >> 1);
    if (gid >= total) return;
    int node = gid / (C >> 1);
    float s  = nd[node];
    float2 a = *(const float2*)(agg + (size_t)gid * 2);
    float2 z = *(const float2*)(h0  + (size_t)gid * 2);
    float2 r;
    r.x = a.x * s + 0.1f * z.x;
    r.y = a.y * s + 0.1f * z.y;
    *(float2*)(hout + (size_t)gid * 2) = r;
    float2 zero = {0.0f, 0.0f};
    *(float2*)(agg + (size_t)gid * 2) = zero;
}

// ---------------------------------------------------------------------------
// Launcher
// ---------------------------------------------------------------------------
extern "C" void kernel_launch(void* const* d_in, const int* in_sizes, int n_in,
                              void* d_out, int out_size, void* d_ws, size_t ws_size,
                              hipStream_t stream)
{
    const float* features = (const float*)d_in[0];
    const int*   edges    = (const int*)d_in[1];
    const float* W0 = (const float*)d_in[2];
    const float* b0 = (const float*)d_in[3];
    const float* W1 = (const float*)d_in[4];
    const float* b1 = (const float*)d_in[5];
    const float* W2 = (const float*)d_in[6];
    const float* b2 = (const float*)d_in[7];

    const int IN = 512;
    const int Nn = in_sizes[0] / IN;       // 100000
    const int E  = in_sizes[1] / 2;        // 1000000
    const int H0 = in_sizes[3];            // 512
    const int H1 = in_sizes[5];            // 256
    const int C  = in_sizes[7];            // 64

    const int* src = edges;
    const int* dst = edges + E;

    // workspace carve (256B aligned)
    char* ws = (char*)d_ws;
    size_t off = 0;
    auto carve = [&](size_t bytes) -> void* {
        void* p = ws + off;
        off += (bytes + 255) & ~(size_t)255;
        return p;
    };
    __bf16* h0b    = (__bf16*)carve((size_t)Nn * H0 * sizeof(__bf16));
    __bf16* h1b    = (__bf16*)carve((size_t)Nn * H1 * sizeof(__bf16));
    float*  logits = (float*) carve((size_t)Nn * C * sizeof(float));
    float*  hbuf   = (float*) carve((size_t)Nn * C * sizeof(float));
    float*  agg    = (float*) carve((size_t)Nn * C * sizeof(float));
    __bf16* w0b    = (__bf16*)carve((size_t)H0 * IN * sizeof(__bf16));
    __bf16* w1b    = (__bf16*)carve((size_t)H1 * H0 * sizeof(__bf16));
    __bf16* w2b    = (__bf16*)carve((size_t)C  * H1 * sizeof(__bf16));
    float*  degO   = (float*) carve((size_t)Nn * sizeof(float));
    float*  degI   = (float*) carve((size_t)Nn * sizeof(float));
    float*  normS  = (float*) carve((size_t)Nn * sizeof(float));
    float*  normD  = (float*) carve((size_t)Nn * sizeof(float));

    // zero accumulators (graph-capturable memsets)
    hipMemsetAsync(agg,  0, (size_t)Nn * C * sizeof(float), stream);
    hipMemsetAsync(degO, 0, (size_t)Nn * sizeof(float), stream);
    hipMemsetAsync(degI, 0, (size_t)Nn * sizeof(float), stream);

    // weights fp32 -> bf16 (one-shot; all three fit in L2 trivially)
    wconvert_kernel<<<(H0 * IN + 255) / 256, 256, 0, stream>>>(W0, w0b, H0 * IN);
    wconvert_kernel<<<(H1 * H0 + 255) / 256, 256, 0, stream>>>(W1, w1b, H1 * H0);
    wconvert_kernel<<<(C  * H1 + 255) / 256, 256, 0, stream>>>(W2, w2b, C  * H1);

    // degrees + norms
    deg_kernel <<<(E  + 255) / 256, 256, 0, stream>>>(src, dst, degO, degI, E);
    norm_kernel<<<(Nn + 255) / 256, 256, 0, stream>>>(degO, degI, normS, normD, Nn);

    // 3-layer MLP on WMMA (bf16 operands, fp32 accumulate)
    dim3 g0((Nn + 127) / 128, H0 / 128);   // NT=8 -> 128 cols per block
    wmma_mlp_layer<8, false, true,  true ><<<g0, 256, 0, stream>>>(features, w0b, b0, h0b, Nn, IN, H0);
    dim3 g1((Nn + 127) / 128, H1 / 128);
    wmma_mlp_layer<8, true,  true,  true ><<<g1, 256, 0, stream>>>(h0b, w1b, b1, h1b, Nn, H0, H1);
    dim3 g2((Nn + 127) / 128, C / 64);     // NT=4 -> 64 cols per block
    wmma_mlp_layer<4, true,  false, false><<<g2, 256, 0, stream>>>(h1b, w2b, b2, logits, Nn, H1, C);

    // APPNP: K=10 steps of scatter-add + update
    const int sBlocks = (E * 32 + 255) / 256;
    const int uBlocks = (Nn * (C / 2) + 255) / 256;
    const float* hcur = logits;   // h starts as h0 = logits
    for (int k = 0; k < 10; ++k) {
        scatter_kernel<<<sBlocks, 256, 0, stream>>>(hcur, src, dst, normS, agg, E, C);
        float* hout = (k == 9) ? (float*)d_out : hbuf;
        update_kernel <<<uBlocks, 256, 0, stream>>>(agg, logits, normD, hout, Nn, C);
        hcur = hbuf;
    }
}
